// SheafAttention_67568425501240
// MI455X (gfx1250) — matrix operand order repaired and checked by reading, not compiled, not run on hardware
//
#include <hip/hip_runtime.h>
#include <hip/hip_bf16.h>

// ---------------------------------------------------------------------------
// Sheaf attention for MI455X (gfx1250, wave32, WMMA f32_16x16x32_f16 + TDM)
// B=4, T=2048, D=1024, H=16, S=64
// ---------------------------------------------------------------------------

typedef __attribute__((ext_vector_type(16))) _Float16 v16h;
typedef __attribute__((ext_vector_type(8)))  _Float16 v8h;
typedef __attribute__((ext_vector_type(8)))  float    v8f;
typedef unsigned int u32x4 __attribute__((ext_vector_type(4)));
typedef int          i32x4 __attribute__((ext_vector_type(4)));
typedef int          i32x8 __attribute__((ext_vector_type(8)));

#define BB 4
#define TT 2048
#define DD 1024
#define HH 16
#define SS 64
#define MTOT (BB * TT)   // 8192

// ---- Tensor Data Mover: 2D f16 tile -> LDS with row padding ---------------
// Pads `pada_code` (0=1dw..3=4dw..) DWORDs after every `padi_code` (0=2dw,
// 1=4dw, 2=8dw, 3=16dw, 4=32dw) DWORDs stored, reproducing the padded LDS
// row strides used by the WMMA fragment loaders.
__device__ __forceinline__ void tdm_load_2d(unsigned lds_byte_addr,
                                            const void* gptr,
                                            unsigned tile_d0_elems,
                                            unsigned tile_d1_rows,
                                            unsigned stride_elems,
                                            unsigned padi_code,
                                            unsigned pada_code) {
  const unsigned long long ga = (unsigned long long)(uintptr_t)gptr;
  u32x4 g0;
  g0.x = 1u;                                   // count=1, user descriptor
  g0.y = lds_byte_addr;                        // D#.lds_addr
  g0.z = (unsigned)ga;                         // global_addr[31:0]
  g0.w = (unsigned)((ga >> 32) & 0x01FFFFFFu)  // global_addr[56:32]
         | (2u << 30);                         // type = 2 ("image")
  const unsigned td0 = stride_elems;           // tensor_dim0
  const unsigned td1 = 1u << 30;               // tensor_dim1 (no OOB clipping)
  i32x8 g1;
  g1[0] = (int)((1u << 16)                     // data_size = 2 bytes
                | (1u << 20)                   // pad_enable
                | (padi_code << 22) | (pada_code << 25));
  g1[1] = (int)((td0 & 0xFFFFu) << 16);                       // dim0[15:0]
  g1[2] = (int)((td0 >> 16) | ((td1 & 0xFFFFu) << 16));       // dim0 hi, dim1 lo
  g1[3] = (int)((td1 >> 16) | (tile_d0_elems << 16));         // dim1 hi, tile_dim0
  g1[4] = (int)(tile_d1_rows & 0xFFFFu);                      // tile_dim1
  g1[5] = (int)stride_elems;                                  // dim0_stride[31:0]
  g1[6] = 0;
  g1[7] = 0;
  const i32x4 z4 = {0, 0, 0, 0};
#if __clang_major__ >= 23
  const i32x8 z8 = {0, 0, 0, 0, 0, 0, 0, 0};
  __builtin_amdgcn_tensor_load_to_lds(g0, g1, z4, z4, z8, 0);
#else
  __builtin_amdgcn_tensor_load_to_lds(g0, g1, z4, z4, 0);
#endif
}

__device__ __forceinline__ unsigned lds_addr_of(const void* p) {
  // flat aperture maps addr[31:0] directly to the LDS byte offset
  return (unsigned)(uintptr_t)p;
}

// ---- WMMA fragment loaders (ISA 7.12.2 layouts) ---------------------------
// A 16x32 f16: lane L -> M = L%16, K(e) = 8*(L/16) + (e<8 ? e : e+8)
__device__ __forceinline__ v16h frag_a(const _Float16* base, int ld, int lane) {
  const int m = lane & 15, g = lane >> 4;
  const _Float16* r = base + m * ld + 8 * g;
  v8h lo = *(const v8h*)(r);        // K = 8g .. 8g+7
  v8h hi = *(const v8h*)(r + 16);   // K = 8g+16 .. 8g+23
  v16h o;
#pragma unroll
  for (int e = 0; e < 8; ++e) { o[e] = lo[e]; o[8 + e] = hi[e]; }
  return o;
}

// B 32x16 f16 (stored row-major as [n][k]): lane L -> N = L%16, K(e) = 8*(L/16) + (e<8 ? e : e+8)
__device__ __forceinline__ v16h frag_b(const _Float16* base, int ld, int lane) {
  const int n = lane & 15, g = lane >> 4;
  const _Float16* r = base + n * ld + 8 * g;
  v8h lo = *(const v8h*)(r);
  v8h hi = *(const v8h*)(r + 16);
  v16h o;
#pragma unroll
  for (int e = 0; e < 8; ++e) { o[e] = lo[e]; o[8 + e] = hi[e]; }
  return o;
}

__device__ __forceinline__ v8f wmma16(v16h a, v16h b, v8f c) {
  return __builtin_amdgcn_wmma_f32_16x16x32_f16(false, a, false, b, (short)0, c,
                                                false, false);
}

// ---- quantization helpers -------------------------------------------------
__global__ void k_zero6(unsigned* s) {
  if (threadIdx.x < 6) s[threadIdx.x] = 0u;
}

__global__ void k_absmax(const float* __restrict__ w, int n,
                         unsigned* __restrict__ slot) {
  float m = 0.f;
  for (int i = blockIdx.x * blockDim.x + threadIdx.x; i < n;
       i += gridDim.x * blockDim.x)
    m = fmaxf(m, fabsf(w[i]));
#pragma unroll
  for (int off = 16; off > 0; off >>= 1) m = fmaxf(m, __shfl_xor(m, off));
  if ((threadIdx.x & 31) == 0) atomicMax(slot, __float_as_uint(m));
}

// int6 symmetric quantize -> f16 (forward value of the STE)
__global__ void k_quant(const float* __restrict__ w, int n,
                        const unsigned* __restrict__ slot,
                        _Float16* __restrict__ o) {
  const int i = blockIdx.x * blockDim.x + threadIdx.x;
  if (i >= n) return;
  const float amax = __uint_as_float(*slot);
  const float scale = amax / 31.0f + 1e-8f;
  float q = rintf(w[i] / scale);                 // round half-to-even (jnp.round)
  q = fminf(fmaxf(q, -31.f), 31.f) * scale;
  o[i] = (_Float16)q;
}

__global__ void k_cvt(const float* __restrict__ x, int n,
                      _Float16* __restrict__ o) {
  const int i = blockIdx.x * blockDim.x + threadIdx.x;
  if (i < n) o[i] = (_Float16)x[i];
}

// Wc[h*S+sp][k] = sum_s stalk[sp][s] * Wbig[h*S+s][k]  (fold stalk map into proj)
__global__ void k_combine(const _Float16* __restrict__ stalk,
                          const _Float16* __restrict__ wbig,
                          _Float16* __restrict__ o) {
  const int idx = blockIdx.x * blockDim.x + threadIdx.x;  // DD*DD
  const int n = idx >> 10, k = idx & (DD - 1);
  const int h = n >> 6, sp = n & (SS - 1);
  const _Float16* srow = stalk + sp * SS;
  const _Float16* wcol = wbig + (size_t)(h * SS) * DD + k;
  float acc = 0.f;
#pragma unroll 8
  for (int s = 0; s < SS; ++s) acc += (float)srow[s] * (float)wcol[(size_t)s * DD];
  o[(size_t)n * DD + k] = (_Float16)acc;
}

// ---- tiled WMMA GEMM with double-buffered TDM staging ---------------------
// C[M=8192][N=1024] = A(f16) @ W(f16)^T, K=1024
// mode 0: write f16 into (B,H,T,S) layout   mode 1: write f32 row-major (out)
__global__ __launch_bounds__(256) void k_gemm(const _Float16* __restrict__ A,
                                              const _Float16* __restrict__ W,
                                              _Float16* __restrict__ outH,
                                              float* __restrict__ outF,
                                              int mode) {
  constexpr int K = DD;
  constexpr int LDA = 40;  // 32 data + 8 pad halfs (TDM pad: 4dw every 16dw)
  __shared__ _Float16 lA[2][128 * LDA];
  __shared__ _Float16 lB[2][64 * LDA];
  const int tid = threadIdx.x, lane = tid & 31, wave = tid >> 5;
  const int m0 = blockIdx.x * 128, n0 = blockIdx.y * 64;
  const int wm = (wave & 3) * 32, wn = (wave >> 2) * 32;  // 4x2 wave grid

  v8f acc[2][2] = {};

  if (wave == 0) {  // prologue: stage k-tile 0 via Tensor Data Mover
    tdm_load_2d(lds_addr_of(&lA[0][0]), A + (size_t)m0 * K, 32, 128, K, 3, 3);
    tdm_load_2d(lds_addr_of(&lB[0][0]), W + (size_t)n0 * K, 32, 64, K, 3, 3);
  }

  for (int kk = 0, it = 0; kk < K; kk += 32, ++it) {
    const int cur = it & 1;
    __syncthreads();  // compute on buffer cur^1 is finished -> safe to overwrite
    if (wave == 0) {
      if (kk + 32 < K) {  // prefetch next k-tile into the other buffer
        tdm_load_2d(lds_addr_of(&lA[cur ^ 1][0]),
                    A + (size_t)m0 * K + kk + 32, 32, 128, K, 3, 3);
        tdm_load_2d(lds_addr_of(&lB[cur ^ 1][0]),
                    W + (size_t)n0 * K + kk + 32, 32, 64, K, 3, 3);
        __builtin_amdgcn_s_wait_tensorcnt(2);  // in-order: current tile done
      } else {
        __builtin_amdgcn_s_wait_tensorcnt(0);
      }
    }
    __syncthreads();  // current buffer visible to all waves

    const _Float16* cA = &lA[cur][0];
    const _Float16* cB = &lB[cur][0];
    v16h af[2], bf[2];
#pragma unroll
    for (int i = 0; i < 2; ++i) af[i] = frag_a(cA + (wm + i * 16) * LDA, LDA, lane);
#pragma unroll
    for (int j = 0; j < 2; ++j) bf[j] = frag_b(cB + (wn + j * 16) * LDA, LDA, lane);
#pragma unroll
    for (int i = 0; i < 2; ++i)
#pragma unroll
      for (int j = 0; j < 2; ++j) acc[i][j] = wmma16(af[i], bf[j], acc[i][j]);
  }

  const int g = lane >> 4, nl = lane & 15;
#pragma unroll
  for (int i = 0; i < 2; ++i)
#pragma unroll
    for (int j = 0; j < 2; ++j)
#pragma unroll
      for (int r = 0; r < 8; ++r) {
        const int m = m0 + wm + i * 16 + r + 8 * g;
        const int n = n0 + wn + j * 16 + nl;
        const float v = acc[i][j][r];
        if (mode == 0) {
          const int b = m >> 11, t = m & (TT - 1);
          const int h = n >> 6, s = n & (SS - 1);
          outH[(((size_t)b * HH + h) * TT + t) * SS + s] = (_Float16)v;
        } else {
          outF[(size_t)m * DD + n] = v;
        }
      }
}

// ---- flash attention with 2-adic bias, TDM-staged K/V tiles ---------------
// grid (B*H, T/64), block 128 (4 waves; wave w handles 16 query rows)
__global__ __launch_bounds__(128) void k_flash(const _Float16* __restrict__ Q,
                                               const _Float16* __restrict__ Kt,
                                               const _Float16* __restrict__ V,
                                               const float* __restrict__ p_scale,
                                               _Float16* __restrict__ O) {
  constexpr int LDR = 72;  // 64 data + 8 pad halfs (TDM pad: 4dw every 32dw)
  constexpr int LDV = 40;
  __shared__ _Float16 kraw[32 * LDR];         // K tile [key][s]
  __shared__ _Float16 vraw[32 * LDR];         // V tile [key][s]
  __shared__ _Float16 vt[SS * LDV];           // V tile transposed [s][key]
  __shared__ _Float16 pbuf[4][16 * LDV];      // per-wave P stage (C -> A layout)
  const int tid = threadIdx.x, lane = tid & 31, wave = tid >> 5;
  const int bh = blockIdx.x, b = bh >> 4, h = bh & 15;
  const int q0 = (blockIdx.y * 4 + wave) * 16;
  const size_t rowbase = (size_t)bh * TT;
  const int g = lane >> 4, nl = lane & 15;
  const float pscl = p_scale[0];
  const float sscale = 0.125f;  // 1/sqrt(64)

  // Q fragments for both 32-wide K-dim steps (held for the whole key sweep)
  v16h aq[2];
#pragma unroll
  for (int ks = 0; ks < 2; ++ks)
    aq[ks] = frag_a(Q + (rowbase + q0) * SS + ks * 32, SS, lane);

  float mrun[8], lrun[8];
#pragma unroll
  for (int r = 0; r < 8; ++r) { mrun[r] = -3.0e38f; lrun[r] = 0.f; }
  v8f acc[4] = {};

  const int ntiles = blockIdx.y * 2 + 2;  // uniform across the block (causal max)
  for (int kt = 0; kt < ntiles; ++kt) {
    __syncthreads();  // previous tile's compute done -> safe to overwrite
    if (wave == 0) {  // TDM: contiguous 32x64 f16 K and V tiles -> LDS
      tdm_load_2d(lds_addr_of(&kraw[0]),
                  Kt + (rowbase + kt * 32) * SS, 64, 32, 64, 4, 3);
      tdm_load_2d(lds_addr_of(&vraw[0]),
                  V + (rowbase + kt * 32) * SS, 64, 32, 64, 4, 3);
      __builtin_amdgcn_s_wait_tensorcnt(0);
    }
    __syncthreads();
    {  // cooperative LDS->LDS transpose: vt[s][key] = vraw[key][s]
      const int key = tid & 31;
      const int s0 = (tid >> 5) * 16;
      const _Float16* src = &vraw[key * LDR + s0];
#pragma unroll
      for (int e = 0; e < 16; ++e) vt[(s0 + e) * LDV + key] = src[e];
    }
    __syncthreads();

    if (kt * 32 <= q0 + 15) {  // this key tile intersects the causal window
      // scores tile 16x32 = Q(16x64) x K^T(64x32) -> 4 WMMA
      v8f sf[2] = {};
#pragma unroll
      for (int sub = 0; sub < 2; ++sub)
#pragma unroll
        for (int ks = 0; ks < 2; ++ks) {
          v16h bk = frag_b(&kraw[(sub * 16) * LDR + ks * 32], LDR, lane);
          sf[sub] = wmma16(aq[ks], bk, sf[sub]);
        }

      // scale + 2-adic bias + causal mask + online softmax
      float p[2][8];
#pragma unroll
      for (int r = 0; r < 8; ++r) {
        const int qi = q0 + r + 8 * g;
#pragma unroll
        for (int sub = 0; sub < 2; ++sub) {
          const int ki = kt * 32 + sub * 16 + nl;
          const int d = qi - ki;
          float sc = sf[sub][r] * sscale;
          if (d < 0) {
            sc = -3.0e38f;  // causal mask
          } else {
            const float bias =
                (d == 0) ? 1.0f
                         : fminf((float)__builtin_ctz(d), 16.f) * 0.0625f;
            sc += pscl * bias;
          }
          sf[sub][r] = sc;
        }
        // row max across the 16 lanes of this half-wave (N dimension)
        float v = fmaxf(sf[0][r], sf[1][r]);
        v = fmaxf(v, __shfl_xor(v, 1));
        v = fmaxf(v, __shfl_xor(v, 2));
        v = fmaxf(v, __shfl_xor(v, 4));
        v = fmaxf(v, __shfl_xor(v, 8));
        const float mnew = fmaxf(mrun[r], v);
        const float corr = __expf(mrun[r] - mnew);
        mrun[r] = mnew;
        const float p0 = __expf(sf[0][r] - mnew);
        const float p1 = __expf(sf[1][r] - mnew);
        p[0][r] = p0; p[1][r] = p1;
        float rs = p0 + p1;
        rs += __shfl_xor(rs, 1);
        rs += __shfl_xor(rs, 2);
        rs += __shfl_xor(rs, 4);
        rs += __shfl_xor(rs, 8);
        lrun[r] = lrun[r] * corr + rs;
#pragma unroll
        for (int f = 0; f < 4; ++f) acc[f][r] *= corr;
      }

      // C-layout P -> LDS -> A-layout fragment (LDS is in-order per wave)
      _Float16* pb = &pbuf[wave][0];
#pragma unroll
      for (int r = 0; r < 8; ++r) {
        const int m = r + 8 * g;
        pb[m * LDV + nl]      = (_Float16)p[0][r];
        pb[m * LDV + 16 + nl] = (_Float16)p[1][r];
      }
      v16h ap = frag_a(pb, LDV, lane);
#pragma unroll
      for (int ns = 0; ns < 4; ++ns) {
        v16h bv = frag_b(&vt[ns * 16 * LDV], LDV, lane);  // B[k=key][n=s]
        acc[ns] = wmma16(ap, bv, acc[ns]);
      }
    }
  }

  // normalize and scatter to (B,T,D) f16
#pragma unroll
  for (int r = 0; r < 8; ++r) {
    const float inv = 1.0f / lrun[r];
    const int t = q0 + r + 8 * g;
    const size_t orow = ((size_t)b * TT + t) * DD + h * SS;
#pragma unroll
    for (int ns = 0; ns < 4; ++ns)
      O[orow + ns * 16 + nl] = (_Float16)(acc[ns][r] * inv);
  }
}

// ---------------------------------------------------------------------------
extern "C" void kernel_launch(void* const* d_in, const int* in_sizes, int n_in,
                              void* d_out, int out_size, void* d_ws,
                              size_t ws_size, hipStream_t stream) {
  const float* x   = (const float*)d_in[0];
  const float* wq  = (const float*)d_in[1];
  const float* wk  = (const float*)d_in[2];
  const float* wv  = (const float*)d_in[3];
  const float* wo  = (const float*)d_in[4];
  const float* wu  = (const float*)d_in[5];
  const float* wsv = (const float*)d_in[6];
  const float* ps  = (const float*)d_in[7];
  float* out = (float*)d_out;

  char* ws = (char*)d_ws;
  size_t off = 0;
  auto take = [&](size_t bytes) {
    char* p = ws + off;
    off = (off + bytes + 255) & ~(size_t)255;
    return p;
  };
  unsigned*  scales = (unsigned*)take(6 * sizeof(unsigned));
  _Float16*  x16    = (_Float16*)take((size_t)MTOT * DD * 2);
  _Float16*  wq16   = (_Float16*)take((size_t)DD * DD * 2);
  _Float16*  wk16   = (_Float16*)take((size_t)DD * DD * 2);
  _Float16*  wv16   = (_Float16*)take((size_t)DD * DD * 2);
  _Float16*  wo16   = (_Float16*)take((size_t)DD * DD * 2);
  _Float16*  wu16   = (_Float16*)take((size_t)SS * SS * 2);
  _Float16*  wsv16  = (_Float16*)take((size_t)SS * SS * 2);
  _Float16*  wcq16  = (_Float16*)take((size_t)DD * DD * 2);   // blockdiag(Wu)@Wq
  _Float16*  wck16  = (_Float16*)take((size_t)DD * DD * 2);   // blockdiag(Wsv)@Wk
  _Float16*  qs16   = (_Float16*)take((size_t)MTOT * DD * 2); // (B,H,T,S)
  _Float16*  ks16   = (_Float16*)take((size_t)MTOT * DD * 2);
  _Float16*  v16    = (_Float16*)take((size_t)MTOT * DD * 2);
  _Float16*  o16    = (_Float16*)take((size_t)MTOT * DD * 2); // (B,T,D)
  (void)ws_size; (void)n_in; (void)in_sizes; (void)out_size;

  const int nDD = DD * DD, nSS = SS * SS, nX = MTOT * DD;

  // 1) quantization scales
  k_zero6<<<1, 32, 0, stream>>>(scales);
  k_absmax<<<64, 256, 0, stream>>>(wq, nDD, scales + 0);
  k_absmax<<<64, 256, 0, stream>>>(wk, nDD, scales + 1);
  k_absmax<<<64, 256, 0, stream>>>(wv, nDD, scales + 2);
  k_absmax<<<64, 256, 0, stream>>>(wo, nDD, scales + 3);
  k_absmax<<<4, 256, 0, stream>>>(wu, nSS, scales + 4);
  k_absmax<<<4, 256, 0, stream>>>(wsv, nSS, scales + 5);

  // 2) quantize -> f16, convert x -> f16
  k_quant<<<(nDD + 255) / 256, 256, 0, stream>>>(wq, nDD, scales + 0, wq16);
  k_quant<<<(nDD + 255) / 256, 256, 0, stream>>>(wk, nDD, scales + 1, wk16);
  k_quant<<<(nDD + 255) / 256, 256, 0, stream>>>(wv, nDD, scales + 2, wv16);
  k_quant<<<(nDD + 255) / 256, 256, 0, stream>>>(wo, nDD, scales + 3, wo16);
  k_quant<<<(nSS + 255) / 256, 256, 0, stream>>>(wu, nSS, scales + 4, wu16);
  k_quant<<<(nSS + 255) / 256, 256, 0, stream>>>(wsv, nSS, scales + 5, wsv16);
  k_cvt<<<(nX + 255) / 256, 256, 0, stream>>>(x, nX, x16);

  // 3) fold stalk maps into projections
  k_combine<<<nDD / 256, 256, 0, stream>>>(wu16, wq16, wcq16);
  k_combine<<<nDD / 256, 256, 0, stream>>>(wsv16, wk16, wck16);

  // 4) WMMA projections -> Qs, Ks, V in (B,H,T,S) f16
  dim3 gg(MTOT / 128, DD / 64);
  k_gemm<<<gg, 256, 0, stream>>>(x16, wcq16, qs16, nullptr, 0);
  k_gemm<<<gg, 256, 0, stream>>>(x16, wck16, ks16, nullptr, 0);
  k_gemm<<<gg, 256, 0, stream>>>(x16, wv16, v16, nullptr, 0);

  // 5) flash attention with p-adic bias
  k_flash<<<dim3(BB * HH, TT / 64), 128, 0, stream>>>(qs16, ks16, v16, ps, o16);

  // 6) output projection -> fp32
  k_gemm<<<gg, 256, 0, stream>>>(o16, wo16, nullptr, out, 1);
}
